// R_54571854463411
// MI455X (gfx1250) — matrix operand order repaired and checked by reading, not compiled, not run on hardware
//
#include <hip/hip_runtime.h>
#include <hip/hip_bf16.h>

// ---------------------------------------------------------------------------
// Problem constants (match reference)
// ---------------------------------------------------------------------------
#define SB   128     // batch
#define NN   256     // nodes
#define FX   5       // raw features
#define HH   32      // H1 == H2
#define H3   128
#define EPSF 1e-7f
#define MROWS 32     // rows of A per block (2 WMMA M-tiles)

typedef __attribute__((ext_vector_type(16))) __bf16 v16bf;
typedef __attribute__((ext_vector_type(8)))  float  v8f;

__device__ __forceinline__ unsigned short f2bf(float f) {
    // round-to-nearest-even fp32 -> bf16
    unsigned int x = __float_as_uint(f);
    x += 0x7FFFu + ((x >> 16) & 1u);
    return (unsigned short)(x >> 16);
}

__device__ __forceinline__ unsigned int pack2bf(float lo, float hi) {
    return (unsigned int)f2bf(lo) | ((unsigned int)f2bf(hi) << 16);
}

__device__ __forceinline__ float lrelu(float v) {
    return v >= 0.f ? v : 0.01f * v;
}

__device__ __forceinline__ float fastrcp(float x) {
    // v_rcp_f32 + one Newton-Raphson step (~full fp32 accuracy for x>0)
    float r = __builtin_amdgcn_rcpf(x);
    return r * (2.0f - x * r);
}

// ---------------------------------------------------------------------------
// Residual path: R[s,i,:] = x_in[s,i,:] @ Wr + br    (fp32, 32 outputs)
// x_in = concat(hprev[finA], x[finB])
// ---------------------------------------------------------------------------
__global__ void __launch_bounds__(256)
lin_nodes_kernel(const float* __restrict__ hprev, int finA,
                 const float* __restrict__ x,     int finB,
                 const float* __restrict__ Wr, const float* __restrict__ br,
                 float* __restrict__ R)
{
    const int s = blockIdx.x;
    const int i = threadIdx.x;
    const int fin = finA + finB;

    __shared__ float Wl[37 * 32];
    __shared__ float bl[32];
    for (int p = threadIdx.x; p < fin * 32; p += 256) Wl[p] = Wr[p];
    if (threadIdx.x < 32) bl[threadIdx.x] = br[threadIdx.x];
    __syncthreads();

    float acc[32];
#pragma unroll
    for (int a = 0; a < 32; ++a) acc[a] = bl[a];
    for (int f = 0; f < finA; ++f) {
        float v = hprev[((size_t)s * NN + i) * 32 + f];
#pragma unroll
        for (int a = 0; a < 32; ++a) acc[a] += v * Wl[f * 32 + a];
    }
    for (int f = 0; f < finB; ++f) {
        float v = x[((size_t)s * NN + i) * FX + f];
#pragma unroll
        for (int a = 0; a < 32; ++a) acc[a] += v * Wl[(finA + f) * 32 + a];
    }
    float* o = R + ((size_t)s * NN + i) * HH;
#pragma unroll
    for (int a = 0; a < 32; ++a) o[a] = acc[a];
}

// ---------------------------------------------------------------------------
// Build WMMA B operand, column-major bf16:
//   Bt[s][c][a][j], a in [0,48):  a<32 : Xe[s,c,j,a] = x_in[s,j,:]@We[c]+be[c]
//                                 a==32: 1.0  (row-sum column -> normalizer)
//                                 a>32 : 0
// ---------------------------------------------------------------------------
__global__ void __launch_bounds__(256)
build_B_kernel(const float* __restrict__ hprev, int finA,
               const float* __restrict__ x,     int finB,
               const float* __restrict__ We, const float* __restrict__ be,
               unsigned short* __restrict__ Bt)
{
    const int c = blockIdx.x;   // 0..3
    const int s = blockIdx.y;   // 0..127
    const int j = threadIdx.x;  // 0..255
    const int fin = finA + finB;

    __shared__ float Wl[37 * 32];
    __shared__ float bl[32];
    for (int p = threadIdx.x; p < fin * 32; p += 256)
        Wl[p] = We[(size_t)c * fin * 32 + p];
    if (threadIdx.x < 32) bl[threadIdx.x] = be[c * 32 + threadIdx.x];
    __syncthreads();

    float acc[32];
#pragma unroll
    for (int a = 0; a < 32; ++a) acc[a] = bl[a];
    for (int f = 0; f < finA; ++f) {
        float v = hprev[((size_t)s * NN + j) * 32 + f];
#pragma unroll
        for (int a = 0; a < 32; ++a) acc[a] += v * Wl[f * 32 + a];
    }
    for (int f = 0; f < finB; ++f) {
        float v = x[((size_t)s * NN + j) * FX + f];
#pragma unroll
        for (int a = 0; a < 32; ++a) acc[a] += v * Wl[(finA + f) * 32 + a];
    }

    unsigned short* base = Bt + (((size_t)s * 4 + c) * 48) * NN + j;
#pragma unroll
    for (int a = 0; a < 32; ++a) base[a * NN] = f2bf(acc[a]);
    base[32 * NN] = 0x3F80;  // bf16 1.0 -> row-sum column
#pragma unroll
    for (int a = 33; a < 48; ++a) base[a * NN] = 0;
}

// ---------------------------------------------------------------------------
// Main conv kernel (WMMA). One block = 3 waves handles 32 rows (i) of one
// batch s for all 4 channels over full K=256:
//   waves 0,1 -> output cols 0..31 ; wave 2 -> ones-column tile (row sums).
// Each wave owns 2 M-tiles (rows 0-15, 16-31) sharing one B fragment per
// K-step. Per channel: msg = Ae(32x256 bf16) x B(256x48 bf16) via
// v_wmma_f32_16x16x32_bf16; norm = 1/(col32 + eps) broadcast through LDS;
// out += msg * norm. Finally out = lrelu(out + R).
// ---------------------------------------------------------------------------
__global__ void __launch_bounds__(96)
conv_main_kernel(const float* __restrict__ A,
                 const unsigned short* __restrict__ Bt,
                 const float* __restrict__ R,
                 float* __restrict__ hout)
{
    const int i0   = blockIdx.x * MROWS;
    const int s    = blockIdx.y;
    const int tid  = threadIdx.x;
    const int w    = tid >> 5;    // wave id (wave32)
    const int lane = tid & 31;
    const int half = lane >> 4;
    const int nl   = lane & 15;

    // [c][m][k] bf16, stored as packed uints (two k per uint) -> 64KB
    __shared__ __align__(16) unsigned int AtileU[4 * MROWS * (NN / 2)];
    __shared__ float normL[MROWS];
    const unsigned short* Atile = (const unsigned short*)AtileU;

    // ---- stream A tile (32 rows x 256 cols x 4ch), fp32 -> packed bf16 ----
    // Two consecutive nodes are 40B apart -> 8B aligned float2 loads.
    const float2* Ab2 = (const float2*)(A + (((size_t)s * NN + i0) * NN) * 5);
    for (int p2 = tid; p2 < MROWS * (NN / 2); p2 += 96) {
        const float2* g = Ab2 + (size_t)p2 * 5;       // p2 = m*128 + jpair
        __builtin_prefetch(g + 96 * 5, 0, 0);         // next loop iteration
        float2 f0 = g[0], f1 = g[1], f2 = g[2], f3 = g[3], f4 = g[4];
        // node0 ch c = float[c], node1 ch c = float[5+c]
        AtileU[0 * 4096 + p2] = pack2bf(f0.x, f2.y);
        AtileU[1 * 4096 + p2] = pack2bf(f0.y, f3.x);
        AtileU[2 * 4096 + p2] = pack2bf(f1.x, f3.y);
        AtileU[3 * 4096 + p2] = pack2bf(f1.y, f4.x);
    }
    __syncthreads();

    union Frag { v16bf v; uint4 u[2]; };

    v8f outacc0 = {}, outacc1 = {};
    const int n0 = w * 16;

    for (int c = 0; c < 4; ++c) {
        v8f acc0 = {}, acc1 = {};
        const unsigned short* Bc =
            Bt + (((size_t)s * 4 + c) * 48 + (n0 + nl)) * NN;
        const unsigned short* Ac = Atile + c * (MROWS * NN);

#pragma unroll
        for (int k0 = 0; k0 < 256; k0 += 32) {
            // B fragment (32x16 bf16): 16 contiguous K per lane; shared by
            // both M-tiles.
            Frag fb;
            const unsigned short* bp = Bc + k0 + half * 16;
            fb.u[0] = *(const uint4*)(bp);
            fb.u[1] = *(const uint4*)(bp + 8);
            // A fragments (16x32 bf16): 16 contiguous K per lane, matching B
            Frag fa0, fa1;
            const unsigned short* ap0 = Ac + nl * NN + k0 + half * 16;
            fa0.u[0] = *(const uint4*)(ap0);
            fa0.u[1] = *(const uint4*)(ap0 + 8);
            const unsigned short* ap1 = ap0 + 16 * NN;
            fa1.u[0] = *(const uint4*)(ap1);
            fa1.u[1] = *(const uint4*)(ap1 + 8);

            acc0 = __builtin_amdgcn_wmma_f32_16x16x32_bf16(
                false, fa0.v, false, fb.v, (short)0, acc0, false, false);
            acc1 = __builtin_amdgcn_wmma_f32_16x16x32_bf16(
                false, fa1.v, false, fb.v, (short)0, acc1, false, false);
        }

        __syncthreads();  // protect normL vs previous channel's readers
        if (w == 2 && nl == 0) {
            // lane 0 holds col 32 for M = v, lane 16 for M = 8+v
#pragma unroll
            for (int v = 0; v < 8; ++v) {
                normL[v + 8 * half]      = fastrcp(acc0[v] + EPSF);
                normL[16 + v + 8 * half] = fastrcp(acc1[v] + EPSF);
            }
        }
        __syncthreads();
        if (w < 2) {
#pragma unroll
            for (int v = 0; v < 8; ++v) {
                outacc0[v] += acc0[v] * normL[v + 8 * half];
                outacc1[v] += acc1[v] * normL[16 + v + 8 * half];
            }
        }
    }

    if (w < 2) {
#pragma unroll
        for (int v = 0; v < 8; ++v) {
            int col = n0 + nl;                 // N = lane & 15
            int row0 = i0 + v + 8 * half;      // D layout: M = v + 8*(lane>>4)
            size_t idx0 = ((size_t)s * NN + row0) * HH + col;
            hout[idx0] = lrelu(outacc0[v] + R[idx0]);
            size_t idx1 = idx0 + (size_t)16 * HH;   // rows 16..31
            hout[idx1] = lrelu(outacc1[v] + R[idx1]);
        }
    }
}

// ---------------------------------------------------------------------------
// Head: gate -> softmax over nodes -> attention-pooled feat -> MLP -> scalar
// One block per batch element.
// ---------------------------------------------------------------------------
__global__ void __launch_bounds__(256)
pool_head_kernel(const float* __restrict__ h2, const float* __restrict__ x,
                 const float* __restrict__ Wg, const float* __restrict__ bg,
                 const float* __restrict__ Wn, const float* __restrict__ bn,
                 const float* __restrict__ W1, const float* __restrict__ b1,
                 const float* __restrict__ W2, const float* __restrict__ b2,
                 const float* __restrict__ W3, const float* __restrict__ b3,
                 float* __restrict__ out)
{
    const int s = blockIdx.x;
    const int t = threadIdx.x;

    __shared__ float hx[NN][37];       // h2x rows
    __shared__ float WnL[37 * H3];
    __shared__ float attn[NN];
    __shared__ float red[NN];
    __shared__ float gv[H3];
    __shared__ float h1m[H3];
    __shared__ float h2m[H3];

    {   // stage h2x row t and Wn
        const float* hr = h2 + ((size_t)s * NN + t) * HH;
#pragma unroll
        for (int f = 0; f < 32; ++f) hx[t][f] = hr[f];
        const float* xr = x + ((size_t)s * NN + t) * FX;
#pragma unroll
        for (int f = 0; f < FX; ++f) hx[t][32 + f] = xr[f];
    }
    for (int p = t; p < 37 * H3; p += 256) WnL[p] = Wn[p];
    __syncthreads();

    // gate + softmax over the 256 nodes
    float gacc = bg[0];
#pragma unroll
    for (int f = 0; f < 37; ++f) gacc += hx[t][f] * Wg[f];
    red[t] = gacc;
    __syncthreads();
    for (int st = 128; st > 0; st >>= 1) {
        if (t < st) red[t] = fmaxf(red[t], red[t + st]);
        __syncthreads();
    }
    float mx = red[0];
    __syncthreads();
    float e = expf(gacc - mx);
    attn[t] = e;
    red[t]  = e;
    __syncthreads();
    for (int st = 128; st > 0; st >>= 1) {
        if (t < st) red[t] += red[t + st];
        __syncthreads();
    }
    float inv = fastrcp(red[0]);
    __syncthreads();
    attn[t] *= inv;
    __syncthreads();

    // feat = lrelu(h2x@Wn+bn); g = lrelu(sum_i attn_i * feat_i)
    if (t < H3) {
        float acc = 0.f;
        for (int i = 0; i < NN; ++i) {
            float f = bn[t];
#pragma unroll 1
            for (int ff = 0; ff < 37; ++ff) f += hx[i][ff] * WnL[ff * H3 + t];
            acc += attn[i] * lrelu(f);
        }
        gv[t] = lrelu(acc);
    }
    __syncthreads();

    if (t < H3) {
        float a1 = b1[t];
        for (int f = 0; f < H3; ++f) a1 += gv[f] * W1[f * H3 + t];
        h1m[t] = lrelu(a1);
    }
    __syncthreads();
    if (t < H3) {
        float a2 = b2[t];
        for (int f = 0; f < H3; ++f) a2 += h1m[f] * W2[f * H3 + t];
        h2m[t] = lrelu(a2);
    }
    __syncthreads();
    if (t < H3) red[t] = h2m[t] * W3[t];
    __syncthreads();
    for (int st = 64; st > 0; st >>= 1) {
        if (t < st) red[t] += red[t + st];
        __syncthreads();
    }
    if (t == 0) out[s] = red[0] + b3[0];
}

// ---------------------------------------------------------------------------
// Host launcher
// ---------------------------------------------------------------------------
extern "C" void kernel_launch(void* const* d_in, const int* in_sizes, int n_in,
                              void* d_out, int out_size, void* d_ws, size_t ws_size,
                              hipStream_t stream)
{
    (void)in_sizes; (void)n_in; (void)out_size; (void)ws_size;

    const float* A    = (const float*)d_in[0];
    const float* x    = (const float*)d_in[1];
    const float* c1We = (const float*)d_in[2];
    const float* c1be = (const float*)d_in[3];
    const float* c1Wr = (const float*)d_in[4];
    const float* c1br = (const float*)d_in[5];
    const float* c2We = (const float*)d_in[6];
    const float* c2be = (const float*)d_in[7];
    const float* c2Wr = (const float*)d_in[8];
    const float* c2br = (const float*)d_in[9];
    const float* Wg   = (const float*)d_in[10];
    const float* bg   = (const float*)d_in[11];
    const float* Wn   = (const float*)d_in[12];
    const float* bn   = (const float*)d_in[13];
    const float* W1   = (const float*)d_in[14];
    const float* b1   = (const float*)d_in[15];
    const float* W2   = (const float*)d_in[16];
    const float* b2   = (const float*)d_in[17];
    const float* W3   = (const float*)d_in[18];
    const float* b3   = (const float*)d_in[19];
    float* out = (float*)d_out;

    // workspace layout
    char* ws = (char*)d_ws;
    const size_t btBytes = (size_t)SB * 4 * 48 * NN * sizeof(unsigned short); // 12.58 MB
    const size_t hBytes  = (size_t)SB * NN * HH * sizeof(float);              // 4.19 MB
    unsigned short* Bt = (unsigned short*)ws;
    float* Rbuf = (float*)(ws + btBytes);
    float* h1   = (float*)(ws + btBytes + hBytes);
    float* h2   = (float*)(ws + btBytes + 2 * hBytes);

    dim3 gB(4, SB), bB(256);
    dim3 gC(NN / MROWS, SB), bC(96);

    // -------- layer 1 --------
    lin_nodes_kernel<<<dim3(SB), 256, 0, stream>>>(nullptr, 0, x, FX, c1Wr, c1br, Rbuf);
    build_B_kernel  <<<gB, bB, 0, stream>>>(nullptr, 0, x, FX, c1We, c1be, Bt);
    conv_main_kernel<<<gC, bC, 0, stream>>>(A, Bt, Rbuf, h1);

    // -------- layer 2 (input = concat(h1, x)) --------
    lin_nodes_kernel<<<dim3(SB), 256, 0, stream>>>(h1, 32, x, FX, c2Wr, c2br, Rbuf);
    build_B_kernel  <<<gB, bB, 0, stream>>>(h1, 32, x, FX, c2We, c2be, Bt);
    conv_main_kernel<<<gC, bC, 0, stream>>>(A, Bt, Rbuf, h2);

    // -------- attention pool + MLP head --------
    pool_head_kernel<<<dim3(SB), 256, 0, stream>>>(h2, x, Wg, bg, Wn, bn,
                                                   W1, b1, W2, b2, W3, b3, out);
}
